// AffinityGNN_19061064860222
// MI455X (gfx1250) — matrix operand order repaired and checked by reading, compiled, not run on hardware
//
#include <hip/hip_runtime.h>
#include <hip/hip_bf16.h>

typedef _Float16 v16h __attribute__((ext_vector_type(16)));
typedef _Float16 v8h  __attribute__((ext_vector_type(8)));
typedef float    v8f  __attribute__((ext_vector_type(8)));

namespace {
constexpr int kHD   = 256;
constexpr int kDPH  = 32;
constexpr int kN    = 10000;
constexpr int kE    = 160000;
constexpr int kNPG  = 1250;
constexpr int kP    = 800;
constexpr int kK1   = 515;   // real K of edge MLP layer-1
constexpr int kK1P  = 544;   // padded to multiple of 32
constexpr int kH1   = 512;   // edge MLP hidden
constexpr float kEps = 1e-5f;
}

// ---------------------------------------------------------------- helpers ---

// tanh-approx GELU via v_exp_f32 + v_rcp_f32 (branch-free)
__device__ __forceinline__ float gelu_fast(float x) {
  const float z = 1.5957691216057308f * x * (1.0f + 0.044715f * x * x);
  return x * __builtin_amdgcn_rcpf(1.0f + __expf(-z));
}

__device__ __forceinline__ float sigmoid_fast(float x) {
  return __builtin_amdgcn_rcpf(1.0f + __expf(-x));
}

// A-fragment, 16-bit A 16x32 (ISA 7.12.2): lane L holds M=L&15;
// halves 0..7 = K[kb..kb+7], halves 8..15 = K[kb+16..kb+23], kb = (L&16)?8:0.
__device__ __forceinline__ v16h load_a_frag(const _Float16* row, int k0, int lane) {
  const int kb = (lane & 16) ? 8 : 0;
  v8h lo = *(const v8h*)(row + k0 + kb);
  v8h hi = *(const v8h*)(row + k0 + kb + 16);
  return __builtin_shufflevector(lo, hi, 0,1,2,3,4,5,6,7,8,9,10,11,12,13,14,15);
}

// B-fragment, 16-bit B 32x16: lane L holds N=L&15; halves 0..7 = K[kb..kb+7],
// halves 8..15 = K[kb+16..kb+23], kb=(L&16)?8:0. Weights stored transposed [N][Kpad].
__device__ __forceinline__ v16h load_b_frag(const _Float16* colrow, int k0, int lane) {
  const int kb = (lane & 16) ? 8 : 0;
  v8h lo = *(const v8h*)(colrow + k0 + kb);
  v8h hi = *(const v8h*)(colrow + k0 + kb + 16);
  return __builtin_shufflevector(lo, hi, 0,1,2,3,4,5,6,7,8,9,10,11,12,13,14,15);
}

__device__ __forceinline__ v8f wmma16(v16h a, v16h b, v8f c) {
  return __builtin_amdgcn_wmma_f32_16x16x32_f16(false, a, false, b, (short)0, c,
                                                false, false);
}

// Async global->LDS 16B copy (CDNA5 GLOBAL_LOAD_ASYNC_TO_LDS_B128, ASYNCcnt).
// lds_off: LDS byte address (= addr[31:0] of a generic pointer to __shared__).
__device__ __forceinline__ void async_ld16(unsigned lds_off, const void* g) {
  asm volatile("global_load_async_to_lds_b128 %0, %1, off"
               :: "v"(lds_off), "v"(g) : "memory");
}
__device__ __forceinline__ void wait_async0() {
  asm volatile("s_wait_asynccnt 0x0" ::: "memory");
}

// ------------------------------------------------------------- pre-passes ---

__global__ void k_zero(float* __restrict__ p, int n) {
  int i = blockIdx.x * 256 + threadIdx.x;
  if (i < n) p[i] = 0.f;
}

// dst[n][k] (f16, Kpad) = src[k][n] (f32, K x Nc), zero-padded in K
__global__ void k_transpose_f16(const float* __restrict__ src,
                                _Float16* __restrict__ dst,
                                int K, int Nc, int Kpad) {
  int idx = blockIdx.x * 256 + threadIdx.x;
  if (idx >= Nc * Kpad) return;
  int n = idx / Kpad, k = idx - n * Kpad;
  dst[idx] = (k < K) ? (_Float16)src[(size_t)k * Nc + n] : (_Float16)0.f;
}

__global__ void k_count(const int* __restrict__ eidx, float* __restrict__ cnt) {
  int i = blockIdx.x * 256 + threadIdx.x;
  if (i < kE) atomicAdd(&cnt[eidx[kE + i]], 1.0f);
}

// ---------------------------------------------------------------- encoder ---

struct EncW { const float *w1, *b1, *g, *be, *w2, *b2; };

__global__ __launch_bounds__(128) void k_encode(
    const float* __restrict__ x, EncW prot, EncW lig,
    const float* __restrict__ uw, const float* __restrict__ ub,
    float* __restrict__ xf, float* __restrict__ ident)
{
  __shared__ float sh[128];
  __shared__ float red[128];
  const int n   = blockIdx.x;
  const int tid = threadIdx.x;
  const int ig  = n % kNPG;
  const bool isProt = ig < kP;
  const EncW e  = isProt ? prot : lig;
  const int din = isProt ? 5 : 10;
  const float* xr = x + (size_t)n * 15 + (isProt ? 0 : 5);

  float h = e.b1[tid];
  for (int k = 0; k < din; ++k) h += xr[k] * e.w1[k * 128 + tid];

  red[tid] = h; __syncthreads();
  for (int s = 64; s > 0; s >>= 1) { if (tid < s) red[tid] += red[tid + s]; __syncthreads(); }
  const float mu = red[0] * (1.f / 128.f);
  __syncthreads();
  const float d = h - mu;
  red[tid] = d * d; __syncthreads();
  for (int s = 64; s > 0; s >>= 1) { if (tid < s) red[tid] += red[tid + s]; __syncthreads(); }
  const float rs = rsqrtf(red[0] * (1.f / 128.f) + kEps);
  __syncthreads();

  sh[tid] = gelu_fast(d * rs * e.g[tid] + e.be[tid]);
  __syncthreads();

  float o = e.b2[tid];
  for (int j = 0; j < 128; ++j) o += sh[j] * e.w2[j * 128 + tid];
  red[tid] = o;
  __syncthreads();

  for (int c = tid; c < kHD; c += 128) {
    float u = ub[c];
    for (int j = 0; j < 128; ++j) u += red[j] * uw[j * kHD + c];
    const size_t off = (size_t)n * kHD + c;
    xf[off] = u; ident[off] = u;
  }
}

// ---------------------------------------------------------- layernorm f16 ---

__global__ __launch_bounds__(128) void k_ln(
    const float* __restrict__ xf, const float* __restrict__ g,
    const float* __restrict__ b, _Float16* __restrict__ xh)
{
  __shared__ float red[128];
  const int n = blockIdx.x, tid = threadIdx.x;
  const float* row = xf + (size_t)n * kHD;
  const float v0 = row[tid], v1 = row[tid + 128];
  red[tid] = v0 + v1; __syncthreads();
  for (int s = 64; s > 0; s >>= 1) { if (tid < s) red[tid] += red[tid + s]; __syncthreads(); }
  const float mu = red[0] * (1.f / (float)kHD);
  __syncthreads();
  const float d0 = v0 - mu, d1 = v1 - mu;
  red[tid] = d0 * d0 + d1 * d1; __syncthreads();
  for (int s = 64; s > 0; s >>= 1) { if (tid < s) red[tid] += red[tid + s]; __syncthreads(); }
  const float rs = rsqrtf(red[0] * (1.f / (float)kHD) + kEps);
  _Float16* out = xh + (size_t)n * kHD;
  out[tid]       = (_Float16)(d0 * rs * g[tid] + b[tid]);
  out[tid + 128] = (_Float16)(d1 * rs * g[tid + 128] + b[tid + 128]);
}

// ------------------------------------------- fused edge Q/K/V + attention ---

struct EdgeW {
  const _Float16* w1t[3];   // [512][544] f16 (transposed, K-padded)
  const float*    b1[3];
  const _Float16* w2t[3];   // [256][512] f16
  const float*    b2[3];
  const float *spw1, *spb1, *spw2, *spb2;
};

__global__ __launch_bounds__(128) void k_edge(
    const _Float16* __restrict__ xh, const int* __restrict__ eidx,
    const float* __restrict__ eattr, EdgeW wp, float* __restrict__ agg)
{
  __shared__ __attribute__((aligned(32))) _Float16 sA[16][kK1P];
  __shared__ __attribute__((aligned(32))) _Float16 sH[16][kH1];
  __shared__ float sQKV[3][16][kHD];    // Q, K, V
  __shared__ float sScr[16][64];        // sp hidden, then scores/attn
  __shared__ float sBias[16][8];
  __shared__ int   sSrc[16], sDst[16];

  const int tid  = threadIdx.x;
  const int lane = tid & 31;
  const int wv   = tid >> 5;
  const int e0   = blockIdx.x << 4;

  if (tid < 16) { sSrc[tid] = eidx[e0 + tid]; sDst[tid] = eidx[kE + e0 + tid]; }
  __syncthreads();

  // gather node rows via async global->LDS (512B per row = 32 lanes x 16B)
  for (int r = wv; r < 32; r += 4) {
    const int m = r & 15;
    const int node = (r < 16) ? sSrc[m] : sDst[m];
    const int coff = (r < 16) ? 0 : kHD;
    const unsigned lds = (unsigned)(uintptr_t)&sA[m][coff + lane * 8];
    async_ld16(lds, xh + (size_t)node * kHD + lane * 8);
  }
  // tail: edge_attr[:,0:3] + zero padding (cols 512..543)
  for (int idx = tid; idx < 16 * 32; idx += 128) {
    const int m = idx >> 5, c = 2 * kHD + (idx & 31);
    sA[m][c] = (c < kK1) ? (_Float16)eattr[(size_t)(e0 + m) * 7 + (c - 2 * kHD)]
                         : (_Float16)0.f;
  }
  wait_async0();
  __syncthreads();

  #pragma unroll 1
  for (int q = 0; q < 3; ++q) {
    // L1: [16,544] @ [544,512] + b -> GELU -> sH
    // four N-tiles per wave iteration: shared A-frag, 4 independent accums
    for (int tq = wv; tq < kH1 / 64; tq += 4) {
      v8f a0 = {}, a1 = {}, a2 = {}, a3 = {};
      const _Float16* arow = &sA[lane & 15][0];
      const _Float16* bb = wp.w1t[q] + (size_t)(tq * 64 + (lane & 15)) * kK1P;
      #pragma unroll 1
      for (int kk = 0; kk < kK1P; kk += 32) {
        v16h a = load_a_frag(arow, kk, lane);
        a0 = wmma16(a, load_b_frag(bb,                       kk, lane), a0);
        a1 = wmma16(a, load_b_frag(bb + (size_t)16 * kK1P,   kk, lane), a1);
        a2 = wmma16(a, load_b_frag(bb + (size_t)32 * kK1P,   kk, lane), a2);
        a3 = wmma16(a, load_b_frag(bb + (size_t)48 * kK1P,   kk, lane), a3);
      }
      const int n0 = tq * 64 + (lane & 15);
      const int mb = (lane >> 4) << 3;
      const float c0 = wp.b1[q][n0],      c1 = wp.b1[q][n0 + 16];
      const float c2 = wp.b1[q][n0 + 32], c3 = wp.b1[q][n0 + 48];
      #pragma unroll
      for (int r = 0; r < 8; ++r) {
        sH[mb + r][n0]      = (_Float16)gelu_fast(a0[r] + c0);
        sH[mb + r][n0 + 16] = (_Float16)gelu_fast(a1[r] + c1);
        sH[mb + r][n0 + 32] = (_Float16)gelu_fast(a2[r] + c2);
        sH[mb + r][n0 + 48] = (_Float16)gelu_fast(a3[r] + c3);
      }
    }
    __syncthreads();
    // L2: [16,512] @ [512,256] + b -> sQKV[q]
    for (int tq = wv; tq < kHD / 64; tq += 4) {
      v8f a0 = {}, a1 = {}, a2 = {}, a3 = {};
      const _Float16* arow = &sH[lane & 15][0];
      const _Float16* bb = wp.w2t[q] + (size_t)(tq * 64 + (lane & 15)) * kH1;
      #pragma unroll 1
      for (int kk = 0; kk < kH1; kk += 32) {
        v16h a = load_a_frag(arow, kk, lane);
        a0 = wmma16(a, load_b_frag(bb,                     kk, lane), a0);
        a1 = wmma16(a, load_b_frag(bb + (size_t)16 * kH1,  kk, lane), a1);
        a2 = wmma16(a, load_b_frag(bb + (size_t)32 * kH1,  kk, lane), a2);
        a3 = wmma16(a, load_b_frag(bb + (size_t)48 * kH1,  kk, lane), a3);
      }
      const int n0 = tq * 64 + (lane & 15);
      const int mb = (lane >> 4) << 3;
      const float c0 = wp.b2[q][n0],      c1 = wp.b2[q][n0 + 16];
      const float c2 = wp.b2[q][n0 + 32], c3 = wp.b2[q][n0 + 48];
      #pragma unroll
      for (int r = 0; r < 8; ++r) {
        sQKV[q][mb + r][n0]      = a0[r] + c0;
        sQKV[q][mb + r][n0 + 16] = a1[r] + c1;
        sQKV[q][mb + r][n0 + 32] = a2[r] + c2;
        sQKV[q][mb + r][n0 + 48] = a3[r] + c3;
      }
    }
    __syncthreads();
  }

  // sp MLP hidden: relu(ea[3:7] @ W1 + b1), 64 per edge
  for (int idx = tid; idx < 16 * 64; idx += 128) {
    const int m = idx >> 6, j = idx & 63;
    float a = wp.spb1[j];
    #pragma unroll
    for (int k = 0; k < 4; ++k)
      a += eattr[(size_t)(e0 + m) * 7 + 3 + k] * wp.spw1[k * 64 + j];
    sScr[m][j] = fmaxf(a, 0.f);
  }
  __syncthreads();
  { // sp bias per (edge, head): 16*8 = 128 == blockDim
    const int m = tid >> 3, h = tid & 7;
    float a = wp.spb2[h];
    for (int j = 0; j < 64; ++j) a += sScr[m][j] * wp.spw2[j * 8 + h];
    sBias[m][h] = a;
  }
  __syncthreads();
  // scores[m][h][g] = Q[m,h,:].K[m,g,:]/sqrt(32) + bias[m,h]
  for (int idx = tid; idx < 16 * 64; idx += 128) {
    const int m = idx >> 6, hg = idx & 63, h = hg >> 3, g = hg & 7;
    float s = 0.f;
    #pragma unroll
    for (int d = 0; d < kDPH; ++d)
      s += sQKV[0][m][h * kDPH + d] * sQKV[1][m][g * kDPH + d];
    sScr[m][hg] = s * 0.17677669529663687f + sBias[m][h];
  }
  __syncthreads();
  { // softmax over h for each (m, g)
    const int m = tid >> 3, g = tid & 7;
    float mx = -1e30f;
    #pragma unroll
    for (int h = 0; h < 8; ++h) mx = fmaxf(mx, sScr[m][h * 8 + g]);
    float ex[8], sum = 0.f;
    #pragma unroll
    for (int h = 0; h < 8; ++h) { ex[h] = __expf(sScr[m][h * 8 + g] - mx); sum += ex[h]; }
    const float inv = __builtin_amdgcn_rcpf(sum);
    #pragma unroll
    for (int h = 0; h < 8; ++h) sScr[m][h * 8 + g] = ex[h] * inv;
  }
  __syncthreads();
  // msg[m][h*32+d] = sum_g attn * V ; scatter-add to dst
  for (int idx = tid; idx < 16 * kHD; idx += 128) {
    const int m = idx >> 8, c = idx & 255, h = c >> 5, d = c & 31;
    float s = 0.f;
    #pragma unroll
    for (int g = 0; g < 8; ++g)
      s += sScr[m][h * 8 + g] * sQKV[2][m][g * kDPH + d];
    atomicAdd(&agg[(size_t)sDst[m] * kHD + c], s);
  }
  (void)wv;
}

// ---------------------------------- fused residual + res-lin + LN + FFN ----

struct NodeW {
  const _Float16 *resT;   // [256][512]
  const _Float16 *f1t;    // [512][256]
  const _Float16 *f2t;    // [256][512]
  const float *resB, *f1b, *f2b, *lng, *lnb;
};

__global__ __launch_bounds__(128) void k_combine(
    float* __restrict__ xf, const float* __restrict__ agg,
    const float* __restrict__ count, const float* __restrict__ ident, NodeW wp)
{
  __shared__ float sX[16][kHD];
  __shared__ __attribute__((aligned(32))) _Float16 sAh[16][2 * kHD];
  __shared__ __attribute__((aligned(32))) _Float16 sHh[16][kH1];
  __shared__ __attribute__((aligned(32))) _Float16 sNh[16][kHD];

  const int tid  = threadIdx.x;
  const int lane = tid & 31;
  const int wv   = tid >> 5;
  const int n0b  = blockIdx.x << 4;

  for (int idx = tid; idx < 16 * kHD; idx += 128) {
    const int m = idx >> 8, c = idx & 255;
    const size_t off = (size_t)(n0b + m) * kHD + c;
    const float x0 = xf[off];
    const float x1 = x0 + agg[off] / fmaxf(count[n0b + m], 1.f);
    sX[m][c] = x1;
    sAh[m][c]       = (_Float16)x1;
    sAh[m][kHD + c] = (_Float16)x0;
  }
  __syncthreads();

  // res: x2 = x1 + fused @ Wres + b  ([16,512] @ [512,256]) — 4 tiles/wave
  {
    v8f a0 = {}, a1 = {}, a2 = {}, a3 = {};
    const _Float16* arow = &sAh[lane & 15][0];
    const _Float16* bb = wp.resT + (size_t)(wv * 64 + (lane & 15)) * (2 * kHD);
    #pragma unroll 1
    for (int kk = 0; kk < 2 * kHD; kk += 32) {
      v16h a = load_a_frag(arow, kk, lane);
      a0 = wmma16(a, load_b_frag(bb,                           kk, lane), a0);
      a1 = wmma16(a, load_b_frag(bb + (size_t)16 * 2 * kHD,    kk, lane), a1);
      a2 = wmma16(a, load_b_frag(bb + (size_t)32 * 2 * kHD,    kk, lane), a2);
      a3 = wmma16(a, load_b_frag(bb + (size_t)48 * 2 * kHD,    kk, lane), a3);
    }
    const int n0 = wv * 64 + (lane & 15);
    const int mb = (lane >> 4) << 3;
    const float c0 = wp.resB[n0],      c1 = wp.resB[n0 + 16];
    const float c2 = wp.resB[n0 + 32], c3 = wp.resB[n0 + 48];
    #pragma unroll
    for (int r = 0; r < 8; ++r) {
      sX[mb + r][n0]      += a0[r] + c0;
      sX[mb + r][n0 + 16] += a1[r] + c1;
      sX[mb + r][n0 + 32] += a2[r] + c2;
      sX[mb + r][n0 + 48] += a3[r] + c3;
    }
  }
  __syncthreads();

  // layernorm each of the 16 rows
  if (tid < 16) {
    float mu = 0.f;
    for (int c = 0; c < kHD; ++c) mu += sX[tid][c];
    mu *= (1.f / (float)kHD);
    float var = 0.f;
    for (int c = 0; c < kHD; ++c) { float d = sX[tid][c] - mu; var += d * d; }
    const float rs = rsqrtf(var * (1.f / (float)kHD) + kEps);
    for (int c = 0; c < kHD; ++c)
      sNh[tid][c] = (_Float16)((sX[tid][c] - mu) * rs * wp.lng[c] + wp.lnb[c]);
  }
  __syncthreads();

  // ffn L1: [16,256] @ [256,512] + b -> GELU   (8 quads, 2 per wave)
  for (int tq = wv; tq < kH1 / 64; tq += 4) {
    v8f a0 = {}, a1 = {}, a2 = {}, a3 = {};
    const _Float16* arow = &sNh[lane & 15][0];
    const _Float16* bb = wp.f1t + (size_t)(tq * 64 + (lane & 15)) * kHD;
    #pragma unroll 1
    for (int kk = 0; kk < kHD; kk += 32) {
      v16h a = load_a_frag(arow, kk, lane);
      a0 = wmma16(a, load_b_frag(bb,                     kk, lane), a0);
      a1 = wmma16(a, load_b_frag(bb + (size_t)16 * kHD,  kk, lane), a1);
      a2 = wmma16(a, load_b_frag(bb + (size_t)32 * kHD,  kk, lane), a2);
      a3 = wmma16(a, load_b_frag(bb + (size_t)48 * kHD,  kk, lane), a3);
    }
    const int n0 = tq * 64 + (lane & 15);
    const int mb = (lane >> 4) << 3;
    const float c0 = wp.f1b[n0],      c1 = wp.f1b[n0 + 16];
    const float c2 = wp.f1b[n0 + 32], c3 = wp.f1b[n0 + 48];
    #pragma unroll
    for (int r = 0; r < 8; ++r) {
      sHh[mb + r][n0]      = (_Float16)gelu_fast(a0[r] + c0);
      sHh[mb + r][n0 + 16] = (_Float16)gelu_fast(a1[r] + c1);
      sHh[mb + r][n0 + 32] = (_Float16)gelu_fast(a2[r] + c2);
      sHh[mb + r][n0 + 48] = (_Float16)gelu_fast(a3[r] + c3);
    }
  }
  __syncthreads();

  // ffn L2 + 0.7/0.3 identity blend, write back in place (4 quads, 1 per wave)
  {
    v8f a0 = {}, a1 = {}, a2 = {}, a3 = {};
    const _Float16* arow = &sHh[lane & 15][0];
    const _Float16* bb = wp.f2t + (size_t)(wv * 64 + (lane & 15)) * kH1;
    #pragma unroll 1
    for (int kk = 0; kk < kH1; kk += 32) {
      v16h a = load_a_frag(arow, kk, lane);
      a0 = wmma16(a, load_b_frag(bb,                     kk, lane), a0);
      a1 = wmma16(a, load_b_frag(bb + (size_t)16 * kH1,  kk, lane), a1);
      a2 = wmma16(a, load_b_frag(bb + (size_t)32 * kH1,  kk, lane), a2);
      a3 = wmma16(a, load_b_frag(bb + (size_t)48 * kH1,  kk, lane), a3);
    }
    const int n0 = wv * 64 + (lane & 15);
    const int mb = (lane >> 4) << 3;
    const float c0 = wp.f2b[n0],      c1 = wp.f2b[n0 + 16];
    const float c2 = wp.f2b[n0 + 32], c3 = wp.f2b[n0 + 48];
    #pragma unroll
    for (int r = 0; r < 8; ++r) {
      const int m = mb + r;
      const size_t base = (size_t)(n0b + m) * kHD;
      const float xa = sX[m][n0]      + a0[r] + c0;
      const float xb = sX[m][n0 + 16] + a1[r] + c1;
      const float xc = sX[m][n0 + 32] + a2[r] + c2;
      const float xd = sX[m][n0 + 48] + a3[r] + c3;
      xf[base + n0]      = 0.7f * xa + 0.3f * ident[base + n0];
      xf[base + n0 + 16] = 0.7f * xb + 0.3f * ident[base + n0 + 16];
      xf[base + n0 + 32] = 0.7f * xc + 0.3f * ident[base + n0 + 32];
      xf[base + n0 + 48] = 0.7f * xd + 0.3f * ident[base + n0 + 48];
    }
  }
}

// --------------------------------------------------------------- readout ---

__global__ __launch_bounds__(256) void k_pool(const float* __restrict__ xf,
                                              float* __restrict__ out) {
  const int b = blockIdx.x, c = threadIdx.x;
  float s = 0.f;
  for (int j = 0; j < kNPG; ++j) s += xf[((size_t)b * kNPG + j) * kHD + c];
  out[b * kHD + c] = s * (1.f / (float)kNPG);
}

__global__ __launch_bounds__(64) void k_contrib(
    const float* __restrict__ pooled, const float* __restrict__ w1,
    const float* __restrict__ b1, const float* __restrict__ w2,
    const float* __restrict__ b2, float* __restrict__ out) {
  __shared__ float hid[64];
  const int b = blockIdx.x, tid = threadIdx.x;
  float a = b1[tid];
  for (int k = 0; k < kHD; ++k) a += pooled[b * kHD + k] * w1[k * 64 + tid];
  hid[tid] = fmaxf(a, 0.f);
  __syncthreads();
  if (tid < 3) {
    float o = b2[tid];
    for (int j = 0; j < 64; ++j) o += hid[j] * w2[j * 3 + tid];
    out[8 * kHD + b * 3 + tid] = sigmoid_fast(o);
  }
}

// ------------------------------------------------------------------ host ---

namespace {
constexpr size_t SZ_XF  = (size_t)kN * kHD * 4;          // 10,240,000
constexpr size_t OFF_XF = 0;
constexpr size_t OFF_ID = SZ_XF;
constexpr size_t OFF_AG = 2 * SZ_XF;
constexpr size_t OFF_XH = 3 * SZ_XF;                      // f16 normed nodes
constexpr size_t OFF_CN = OFF_XH + (size_t)kN * kHD * 2;  // degree counts
constexpr size_t OFF_W  = OFF_CN + 40192;                 // f16 weights
constexpr size_t SZ_W1T = (size_t)512 * 544 * 2;
constexpr size_t SZ_W2T = (size_t)256 * 512 * 2;
constexpr size_t OFW_RES = 3 * SZ_W1T + 3 * SZ_W2T;
constexpr size_t OFW_F1  = OFW_RES + SZ_W2T;
constexpr size_t OFW_F2  = OFW_F1 + (size_t)512 * 256 * 2;
constexpr size_t LSTR    = OFW_F2 + SZ_W2T;

// flattened input indices (setup_inputs() insertion order, recursive)
constexpr int I_LAYER0 = 17, I_LSTRIDE = 24;

void tr(const float* src, _Float16* dst, int K, int Nc, int Kpad, hipStream_t s) {
  const int total = Nc * Kpad;
  k_transpose_f16<<<(total + 255) / 256, 256, 0, s>>>(src, dst, K, Nc, Kpad);
}
}

extern "C" void kernel_launch(void* const* d_in, const int* in_sizes, int n_in,
                              void* d_out, int out_size, void* d_ws, size_t ws_size,
                              hipStream_t stream) {
  (void)in_sizes; (void)n_in; (void)out_size; (void)ws_size;
  const float* x     = (const float*)d_in[0];
  const int*   eidx  = (const int*)d_in[1];
  const float* eattr = (const float*)d_in[2];
  auto F = [&](int i) { return (const float*)d_in[i]; };

  char* ws = (char*)d_ws;
  float*    xf    = (float*)(ws + OFF_XF);
  float*    ident = (float*)(ws + OFF_ID);
  float*    agg   = (float*)(ws + OFF_AG);
  _Float16* xh    = (_Float16*)(ws + OFF_XH);
  float*    cnt   = (float*)(ws + OFF_CN);

  // one-time: transpose+convert weights to f16 (deterministic every call)
  for (int L = 0; L < 6; ++L) {
    const int Li = I_LAYER0 + I_LSTRIDE * L;
    char* wb = ws + OFF_W + (size_t)L * LSTR;
    for (int q = 0; q < 3; ++q) {
      tr(F(Li + 4 * q + 0), (_Float16*)(wb + q * SZ_W1T), 515, 512, 544, stream);
      tr(F(Li + 4 * q + 2), (_Float16*)(wb + 3 * SZ_W1T + q * SZ_W2T), 512, 256, 512, stream);
    }
    tr(F(Li + 22), (_Float16*)(wb + OFW_RES), 512, 256, 512, stream);
    tr(F(Li + 18), (_Float16*)(wb + OFW_F1), 256, 512, 256, stream);
    tr(F(Li + 20), (_Float16*)(wb + OFW_F2), 512, 256, 512, stream);
  }

  // degree counts (layer-invariant)
  k_zero<<<(kN + 255) / 256, 256, 0, stream>>>(cnt, kN);
  k_count<<<(kE + 255) / 256, 256, 0, stream>>>(eidx, cnt);

  // encoders + unify -> xf, identity
  EncW prot{F(3), F(4), F(5), F(6), F(7), F(8)};
  EncW lig {F(9), F(10), F(11), F(12), F(13), F(14)};
  k_encode<<<kN, 128, 0, stream>>>(x, prot, lig, F(15), F(16), xf, ident);

  for (int L = 0; L < 6; ++L) {
    const int Li = I_LAYER0 + I_LSTRIDE * L;
    char* wb = ws + OFF_W + (size_t)L * LSTR;

    k_ln<<<kN, 128, 0, stream>>>(xf, F(Li + 16), F(Li + 17), xh);
    k_zero<<<(kN * kHD + 255) / 256, 256, 0, stream>>>(agg, kN * kHD);

    EdgeW ew;
    for (int q = 0; q < 3; ++q) {
      ew.w1t[q] = (const _Float16*)(wb + q * SZ_W1T);
      ew.b1[q]  = F(Li + 4 * q + 1);
      ew.w2t[q] = (const _Float16*)(wb + 3 * SZ_W1T + q * SZ_W2T);
      ew.b2[q]  = F(Li + 4 * q + 3);
    }
    ew.spw1 = F(Li + 12); ew.spb1 = F(Li + 13);
    ew.spw2 = F(Li + 14); ew.spb2 = F(Li + 15);
    k_edge<<<kE / 16, 128, 0, stream>>>(xh, eidx, eattr, ew, agg);

    NodeW nw{ (const _Float16*)(wb + OFW_RES),
              (const _Float16*)(wb + OFW_F1),
              (const _Float16*)(wb + OFW_F2),
              F(Li + 23), F(Li + 19), F(Li + 21), F(Li + 16), F(Li + 17) };
    k_combine<<<kN / 16, 128, 0, stream>>>(xf, agg, cnt, ident, nw);
  }

  float* out = (float*)d_out;
  k_pool<<<8, 256, 0, stream>>>(xf, out);
  k_contrib<<<8, 64, 0, stream>>>(out, F(161), F(162), F(163), F(164), out);
}
